// SpikingNeuralNetwork_80418967650913
// MI455X (gfx1250) — compile-verified, matrix operand order checked by reading, not run on hardware
//
#include <hip/hip_runtime.h>
#include <hip/hip_bf16.h>

// ---------------------------------------------------------------------------
// SNN forward: 3x WMMA-f16 GEMM (fp32 accum) with fused sigmoid / Hodgkin-
// Huxley / spike-detect epilogues, plus elementwise STDP weight update.
// N = 2048 everywhere; all tile sizes divide N exactly -> no bounds checks.
// Double-buffered LDS staging, one barrier per K-step. Issue order per step:
// all fragment ds_loads first, then next-chunk global prefetch, then WMMAs,
// then convert+ds_store -- so the loadcnt wait lands after the matrix math.
// ---------------------------------------------------------------------------

#define NN 2048
#define LDSK 40           // padded LDS row stride in halves (80B -> bank-conflict free)

typedef __attribute__((ext_vector_type(16))) _Float16 v16h;
typedef __attribute__((ext_vector_type(8)))  _Float16 v8h;
typedef __attribute__((ext_vector_type(4)))  _Float16 v4h;
typedef __attribute__((ext_vector_type(8)))  float    v8f;
typedef __attribute__((ext_vector_type(4)))  float    v4f;   // 16B, lowers to b128

__device__ __forceinline__ float fsigmoid(float z) {
    return 1.0f / (1.0f + __expf(-z));
}

// ---- staging: global -> registers -> LDS (f16), split for pipelining ------

struct RegsF32 { v4f f[4]; };   // 128x32 f32 tile, 4 x 16B per thread
struct RegsF16 { v8h h[2]; };   // 128x32 f16 tile, 2 x 16B per thread

__device__ __forceinline__ void gload_f32(const float* __restrict__ srcu, RegsF32& s) {
    const int t = threadIdx.x;
    const unsigned off = (unsigned)((t >> 3) * NN + (t & 7) * 4);  // 32-bit lane offset
#pragma unroll
    for (int p = 0; p < 4; ++p)
        s.f[p] = *(const v4f*)(srcu + off + p * 32 * NN);          // +p*256KB -> IOFFSET
}

__device__ __forceinline__ void lstore_f32(const RegsF32& s, _Float16* __restrict__ dst) {
    const int t = threadIdx.x;
    const int chunk = t & 7;
    const int r0 = t >> 3;
#pragma unroll
    for (int p = 0; p < 4; ++p) {
        v4h h;
        h[0] = (_Float16)s.f[p][0]; h[1] = (_Float16)s.f[p][1];
        h[2] = (_Float16)s.f[p][2]; h[3] = (_Float16)s.f[p][3];
        *(v4h*)(dst + (r0 + p * 32) * LDSK + chunk * 4) = h;   // 8B-aligned ds write
    }
}

__device__ __forceinline__ void gload_f16(const _Float16* __restrict__ srcu, RegsF16& s) {
    const int t = threadIdx.x;
    const unsigned off = (unsigned)((t >> 2) * NN + (t & 3) * 8);  // 32-bit lane offset
#pragma unroll
    for (int p = 0; p < 2; ++p)
        s.h[p] = *(const v8h*)(srcu + off + p * 64 * NN);          // +p*256KB -> IOFFSET
}

__device__ __forceinline__ void lstore_f16(const RegsF16& s, _Float16* __restrict__ dst) {
    const int t = threadIdx.x;
    const int chunk = t & 3;
    const int r0 = t >> 2;
#pragma unroll
    for (int p = 0; p < 2; ++p)
        *(v8h*)(dst + (r0 + p * 64) * LDSK + chunk * 8) = s.h[p];  // 16B-aligned ds write
}

// Load one 16x32 f16 WMMA operand fragment from a staged LDS tile.
// ISA layout (CDNA5 ISA): lane L -> row (L&15); lanes 0-15 hold K={0..7,16..23},
// lanes 16-31 hold K={8..15,24..31}.  Two ds_read_b128 per fragment; the
// 80B row stride makes the 16 row-starts cover all 64 banks exactly once.
__device__ __forceinline__ v16h load_frag(const _Float16* __restrict__ tile, int tilerow) {
    const int l  = threadIdx.x & 31;
    const int r  = tilerow + (l & 15);
    const int hs = l >> 4;
    const _Float16* p = tile + r * LDSK + hs * 8;
    v8h lo = *(const v8h*)p;          // K = hs*8 .. +7
    v8h hi = *(const v8h*)(p + 16);   // K = hs*8+16 .. +23
    v16h f;
#pragma unroll
    for (int i = 0; i < 8; ++i) { f[i] = lo[i]; f[i + 8] = hi[i]; }
    return f;
}

// MODE 1: Z = x@W1^T + b1 -> sigmoid -> membrane/spike -> HH gates (n,m,h) + v(f16)
// MODE 2: Z = v@W2^T + b2 -> sigmoid -> hidden (f16)
// MODE 3: Z = h@W3^T + b3 -> sigmoid -> out (f32)
template <int MODE, bool A_IS_HALF>
__global__ __launch_bounds__(256) void snn_gemm_kernel(
    const void*  __restrict__ Ap,      // [NN,NN] row-major (f32 or f16)
    const float* __restrict__ Bp,      // [NN,NN] row-major weights (multiplied as B^T)
    const float* __restrict__ bias,    // [NN]
    const float* __restrict__ mp,      // membrane_potential [NN]   (MODE 1)
    const float* __restrict__ nvec,    // n [NN]                    (MODE 1)
    const float* __restrict__ mvec,    // m [NN]                    (MODE 1)
    const float* __restrict__ hvec,    // h [NN]                    (MODE 1)
    _Float16*    __restrict__ outh,    // f16 activation out (MODE 1: v, MODE 2: hidden)
    float*       __restrict__ o_n,     // n_new [NN,NN] (MODE 1) / out (MODE 3)
    float*       __restrict__ o_m,     // m_new [NN,NN] (MODE 1)
    float*       __restrict__ o_h,     // h_new [NN,NN] (MODE 1)
    int*         __restrict__ flags)   // row_spiked [NN]           (MODE 1)
{
    __shared__ _Float16 lA[2][128 * LDSK];   // double buffered: 2 x 10 KB
    __shared__ _Float16 lB[2][128 * LDSK];

    const int Mblk = blockIdx.y * 128;
    const int Nblk = blockIdx.x * 128;
    const int w  = threadIdx.x >> 5;         // wave 0..7
    const int wm = (w & 3) * 32;             // 4 waves across M (32 rows each)
    const int wn = (w >> 2) * 64;            // 2 waves across N (64 cols each)

    // uniform tile bases (stay in SGPRs; K advance is SALU)
    const _Float16* Ah = (const _Float16*)Ap + (size_t)Mblk * NN;
    const float*    Af = (const float*)Ap    + (size_t)Mblk * NN;
    const float*    Bf = Bp                  + (size_t)Nblk * NN;

    v8f acc[2][4] = {};

    RegsF32 ra32, rb;
    RegsF16 ra16;

    // prologue: stage K-chunk 0 into buffer 0
    if constexpr (A_IS_HALF) gload_f16(Ah, ra16);
    else                     gload_f32(Af, ra32);
    gload_f32(Bf, rb);
    if constexpr (A_IS_HALF) lstore_f16(ra16, lA[0]);
    else                     lstore_f32(ra32, lA[0]);
    lstore_f32(rb, lB[0]);

    const int KSTEPS = NN / 32;              // 64
    for (int k = 0; k < KSTEPS; ++k) {
        const int cur = k & 1;
        const int nxt = cur ^ 1;
        const bool more = (k + 1) < KSTEPS;

        __syncthreads();                     // buffer `cur` fully staged

        // 1) issue ALL fragment ds_loads first (pins their dest VGPRs so the
        //    prefetch address temps below cannot alias them -> no early wait)
        v16h a0 = load_frag(lA[cur], wm);
        v16h a1 = load_frag(lA[cur], wm + 16);
        v16h b0 = load_frag(lB[cur], wn);
        v16h b1 = load_frag(lB[cur], wn + 16);
        v16h b2 = load_frag(lB[cur], wn + 32);
        v16h b3 = load_frag(lB[cur], wn + 48);

        // 2) issue next K-chunk global prefetch (in flight across the WMMAs)
        if (more) {
            const int k0 = (k + 1) * 32;     // uniform -> SGPR base bump
            if constexpr (A_IS_HALF) gload_f16(Ah + k0, ra16);
            else                     gload_f32(Af + k0, ra32);
            gload_f32(Bf + k0, rb);
        }

        // 3) matrix math (only dscnt waits needed here)
        acc[0][0] = __builtin_amdgcn_wmma_f32_16x16x32_f16(false, a0, false, b0, (short)0, acc[0][0], false, false);
        acc[1][0] = __builtin_amdgcn_wmma_f32_16x16x32_f16(false, a1, false, b0, (short)0, acc[1][0], false, false);
        acc[0][1] = __builtin_amdgcn_wmma_f32_16x16x32_f16(false, a0, false, b1, (short)0, acc[0][1], false, false);
        acc[1][1] = __builtin_amdgcn_wmma_f32_16x16x32_f16(false, a1, false, b1, (short)0, acc[1][1], false, false);
        acc[0][2] = __builtin_amdgcn_wmma_f32_16x16x32_f16(false, a0, false, b2, (short)0, acc[0][2], false, false);
        acc[1][2] = __builtin_amdgcn_wmma_f32_16x16x32_f16(false, a1, false, b2, (short)0, acc[1][2], false, false);
        acc[0][3] = __builtin_amdgcn_wmma_f32_16x16x32_f16(false, a0, false, b3, (short)0, acc[0][3], false, false);
        acc[1][3] = __builtin_amdgcn_wmma_f32_16x16x32_f16(false, a1, false, b3, (short)0, acc[1][3], false, false);

        // 4) convert + write prefetched chunk into the other buffer
        //    (loadcnt wait lands here, after the WMMA block)
        if (more) {
            if constexpr (A_IS_HALF) lstore_f16(ra16, lA[nxt]);
            else                     lstore_f32(ra32, lA[nxt]);
            lstore_f32(rb, lB[nxt]);
        }
    }

    // ---- fused epilogue (fully unrolled: keep accumulators in VGPRs) ----
    // C/D layout: VGPR e -> M = e + 8*(lane>=16), N = lane&15.
    const int l      = threadIdx.x & 31;
    const int colin  = l & 15;
    const int rowoff = (l >> 4) * 8;

#pragma unroll
    for (int mi = 0; mi < 2; ++mi) {
#pragma unroll
        for (int ni = 0; ni < 4; ++ni) {
#pragma unroll
            for (int e = 0; e < 8; ++e) {
                const int gm = Mblk + wm + mi * 16 + rowoff + e;  // batch row
                const int gn = Nblk + wn + ni * 16 + colin;       // neuron col
                const int idx = gm * NN + gn;
                const float z   = acc[mi][ni][e] + bias[gn];
                const float sig = fsigmoid(z);

                if constexpr (MODE == 1) {
                    float v = 0.5f * mp[gn] + sig;
                    const bool spk = v > 1.0f;
                    if (spk) { atomicOr(&flags[gm], 1); v = 0.0f; }
                    const float an = 0.01f * (v + 55.0f) /
                                     (1.0f - __expf(-(v + 55.0f) * 0.1f));
                    const float bn = 0.125f * __expf(-(v + 65.0f) * (1.0f / 80.0f));
                    const float am = 0.1f * (v + 40.0f) /
                                     (1.0f - __expf(-(v + 40.0f) * 0.1f));
                    const float bm = 4.0f * __expf(-(v + 65.0f) * (1.0f / 18.0f));
                    const float ah = 0.07f * __expf(-(v + 65.0f) * 0.05f);
                    const float bh = fsigmoid((v + 35.0f) * 0.1f);
                    const float n0 = nvec[gn], m0 = mvec[gn], h0 = hvec[gn];
                    o_n[idx] = n0 + 0.01f * (an * (1.0f - n0) - bn * n0);
                    o_m[idx] = m0 + 0.01f * (am * (1.0f - m0) - bm * m0);
                    o_h[idx] = h0 + 0.01f * (ah * (1.0f - h0) - bh * h0);
                    outh[idx] = (_Float16)v;
                } else if constexpr (MODE == 2) {
                    outh[idx] = (_Float16)sig;
                } else {
                    o_n[idx] = sig;   // final network output
                }
            }
        }
    }
}

__global__ void zero_flags_kernel(int* __restrict__ flags) {
    flags[blockIdx.x * 256 + threadIdx.x] = 0;
}

// All-pairs STDP: new_w[i,j] = clip(w[i,j] + (spk_i|spk_j ? dw(ts_i-ts_j) : 0), -1, 1)
// ts = 10 - last_spike_time  =>  dt_mat[i,j] = lst[j] - lst[i].
__global__ __launch_bounds__(256) void stdp_kernel(
    const float* __restrict__ weights, const float* __restrict__ lst,
    const int* __restrict__ flags, float* __restrict__ outw)
{
    const int g  = blockIdx.x * 256 + threadIdx.x;  // one v4f per thread
    const int i  = g >> 9;                          // 512 groups of 4 per row
    const int j4 = (g & 511) * 4;
    const float li = lst[i];
    const int   fi = flags[i];
    const v4f   lj = *(const v4f*)(lst + j4);
    const v4f   wv = *(const v4f*)(weights + (size_t)i * NN + j4);
    v4f res;
#pragma unroll
    for (int t = 0; t < 4; ++t) {
        const float d  = lj[t] - li;
        const float ad = fabsf(d);
        const float dw = (d > 0.0f) ? (-0.015f * __expf(-ad * (1.0f / 25.0f)))
                                    : ( 0.02f  * __expf(-ad * (1.0f / 15.0f)));
        const bool  msk = (fi | flags[j4 + t]) != 0;
        const float wn  = wv[t] + (msk ? dw : 0.0f);
        res[t] = fminf(1.0f, fmaxf(-1.0f, wn));
    }
    *(v4f*)(outw + (size_t)i * NN + j4) = res;
}

extern "C" void kernel_launch(void* const* d_in, const int* in_sizes, int n_in,
                              void* d_out, int out_size, void* d_ws, size_t ws_size,
                              hipStream_t stream) {
    (void)in_sizes; (void)n_in; (void)out_size; (void)ws_size;
    // setup_inputs() order:
    const float* x    = (const float*)d_in[0];
    const float* W1   = (const float*)d_in[1];
    const float* b1   = (const float*)d_in[2];
    const float* W2   = (const float*)d_in[3];
    const float* b2   = (const float*)d_in[4];
    const float* W3   = (const float*)d_in[5];
    const float* b3   = (const float*)d_in[6];
    const float* wts  = (const float*)d_in[7];
    const float* lst  = (const float*)d_in[8];
    const float* mp   = (const float*)d_in[9];
    const float* nv   = (const float*)d_in[10];
    const float* mv   = (const float*)d_in[11];
    const float* hv   = (const float*)d_in[12];

    const size_t NN2 = (size_t)NN * NN;          // 4194304
    float* out   = (float*)d_out;                // [NN,NN] output
    float* out_w = out + NN2;                    // new_weights
    float* out_n = out + 2 * NN2;                // n_new
    float* out_m = out + 3 * NN2;                // m_new
    float* out_h = out + 4 * NN2;                // h_new

    char* ws = (char*)d_ws;
    _Float16* vh    = (_Float16*)ws;                       // 8 MB: v (f16)
    _Float16* hh    = (_Float16*)(ws + (8u << 20));        // 8 MB: hidden (f16)
    int*      flags = (int*)(ws + (16u << 20));            // 8 KB: row_spiked

    dim3 grid(NN / 128, NN / 128), block(256);

    zero_flags_kernel<<<8, 256, 0, stream>>>(flags);

    snn_gemm_kernel<1, false><<<grid, block, 0, stream>>>(
        x, W1, b1, mp, nv, mv, hv, vh, out_n, out_m, out_h, flags);

    snn_gemm_kernel<2, true><<<grid, block, 0, stream>>>(
        vh, W2, b2, nullptr, nullptr, nullptr, nullptr, hh,
        nullptr, nullptr, nullptr, nullptr);

    snn_gemm_kernel<3, true><<<grid, block, 0, stream>>>(
        hh, W3, b3, nullptr, nullptr, nullptr, nullptr, nullptr,
        out, nullptr, nullptr, nullptr);

    stdp_kernel<<<(NN * (NN / 4)) / 256, 256, 0, stream>>>(wts, lst, flags, out_w);
}
